// ConvDeconv_40415642255627
// MI455X (gfx1250) — matrix operand order repaired and checked by reading, not compiled, not run on hardware
//
#include <hip/hip_runtime.h>

// ---------------------------------------------------------------------------
// Sparse conv (stride-2) + transposed conv, MinkowskiEngine kernel-map style.
// MI455X / gfx1250: wave32, WMMA bf16 16x16x32 -> f32 accumulate.
//
// ~18 GFLOP total; feats/y/z fit in the 192MB L2 so HBM traffic is ~300MB
// => ~13us memory bound; the L2 atomic scatter dominates. All f32<->bf16
// conversion is hoisted into one-shot prep kernels; pad map entries resolve
// to dedicated zero rows so the GEMM hot path has no divergent branches:
// vector loads -> v_wmma x16 -> global_atomic_add_f32.
// ---------------------------------------------------------------------------

#define CIN   64
#define COUT  128
#define MAXN2 (64 * 64 * 64)   // upper bound on n_out (stride-2 lattice of 128^3)

typedef __attribute__((ext_vector_type(16))) __bf16 v16bf;
typedef __attribute__((ext_vector_type(8)))  float  v8f;

union Frag {
    v16bf v;
    unsigned int u[8];
};

// f32 -> bf16 via hardware convert (RNE), returned as raw bits
__device__ __forceinline__ unsigned short f2bf(float x) {
    union { __bf16 h; unsigned short s; } u;
    u.h = (__bf16)x;
    return u.s;
}

// --------------------------- prep / utility kernels -------------------------

__global__ void zero_ws_kernel(float* __restrict__ p, long long n) {
    long long i = (long long)blockIdx.x * blockDim.x + threadIdx.x;
    if (i < n) p[i] = 0.0f;
}

// W [27][K][N] f32 -> Wt [27][N][K] bf16 (transposed so B fragments are
// contiguous 16-element runs of K for a fixed output column).
__global__ void cvt_w_kernel(const float* __restrict__ W,
                             unsigned short* __restrict__ Wt, int K, int N) {
    long long i = (long long)blockIdx.x * blockDim.x + threadIdx.x;
    long long tot = 27LL * K * N;
    if (i < tot) {
        int kn   = (int)(i % (K * N));
        int koff = (int)(i / (K * N));
        int kk = kn / N;
        int n  = kn % N;
        Wt[(long long)koff * K * N + (long long)n * K + kk] = f2bf(W[i]);
    }
}

// feats f32 -> bf16 mirror [n_in+1][64]; row n_in is the zero pad row.
__global__ void cvt_feats_kernel(const float* __restrict__ f,
                                 unsigned short* __restrict__ fb, long long n,
                                 long long ntot) {
    long long i = (long long)blockIdx.x * blockDim.x + threadIdx.x;
    if (i < n)          fb[i] = f2bf(f[i]);
    else if (i < ntot)  fb[i] = 0;   // zero pad row
}

// ybf = bf16(y + b1) for rows < n_out; row n_out (pad/trash) zeroed.
__global__ void cvt_y_bias_kernel(const float* __restrict__ y,
                                  const float* __restrict__ b1,
                                  unsigned short* __restrict__ ybf,
                                  const int* __restrict__ n_out_p) {
    const int NOUT = *n_out_p;
    long long i = (long long)blockIdx.x * blockDim.x + threadIdx.x;
    long long lim = (long long)NOUT * COUT;
    if (i < lim)               ybf[i] = f2bf(y[i] + b1[(int)(i & (COUT - 1))]);
    else if (i < lim + COUT)   ybf[i] = 0;   // zero pad row
}

__global__ void init_z_kernel(float* __restrict__ z, const float* __restrict__ b2,
                              long long n) {
    long long i = (long long)blockIdx.x * blockDim.x + threadIdx.x;
    if (i < n) z[i] = b2[(int)(i & (CIN - 1))];
}

// --------------------------- pass 1: sparse conv ----------------------------
// y[out_map[k,m], :] += feats[in_map[k,m], :] @ W1[k]   (64 -> 128 channels)
// Block: 128 threads (4 waves); one offset k, 64 map rows per block.
// Each wave: 16 rows x 128 cols = 8 WMMA N-tiles, K=64 in 2 chunks of 32.

__global__ __launch_bounds__(128) void conv1_kernel(
    const unsigned short* __restrict__ fb,   // feats bf16 [n_in+1][64], last row 0
    const unsigned short* __restrict__ Wt,   // W1 bf16 transposed [27][128][64]
    const int* __restrict__ in_map, const int* __restrict__ out_map,
    const int* __restrict__ n_out_p, float* __restrict__ y, int M, int n_in) {
    __shared__ unsigned short wlds[CIN * COUT];   // [n (0..127)][kk (0..63)]

    const int k   = blockIdx.y;
    const int tid = threadIdx.x;

    // Stage W1t[k] (16KB bf16, already transposed) with vector copies.
    {
        const uint4* src = (const uint4*)(Wt + (long long)k * (CIN * COUT));
        uint4* dst = (uint4*)wlds;
#pragma unroll
        for (int i = 0; i < 8; ++i) dst[tid + 128 * i] = src[tid + 128 * i];
    }
    __syncthreads();

    const int NOUT = *n_out_p;
    const int wave = tid >> 5;
    const int lane = tid & 31;
    const int half = lane >> 4;   // 16-lane group
    const int lidx = lane & 15;
    const int m0   = blockIdx.x * 64 + wave * 16;
    const long long mapbase = (long long)k * M;

    // Scatter/gather indices. Fast path (all but the 27 tail blocks) is fully
    // convergent: plain loads, no exec masking. Pad entries point at zero rows.
    int grow;
    int orow[8];
    if ((blockIdx.x + 1) * 64 <= (unsigned)M) {
        grow = in_map[mapbase + m0 + lidx];
#pragma unroll
        for (int v = 0; v < 8; ++v) orow[v] = out_map[mapbase + m0 + half * 8 + v];
    } else {
        const int m = m0 + lidx;
        grow = (m < M) ? in_map[mapbase + m] : n_in;      // n_in = zero row
#pragma unroll
        for (int v = 0; v < 8; ++v) {
            int mm = m0 + half * 8 + v;
            orow[v] = (mm < M) ? out_map[mapbase + mm] : NOUT;  // NOUT = trash row
        }
    }

    // B fragments: lane holds column n = ntile*16+lidx,
    // K = kchunk*32 + half*16 + {0..15}  (contiguous k order, same as A)
    Frag bf[8][2];
#pragma unroll
    for (int n = 0; n < 8; ++n) {
#pragma unroll
        for (int c = 0; c < 2; ++c) {
            const uint4* p =
                (const uint4*)&wlds[(n * 16 + lidx) * CIN + c * 32 + half * 16];
            uint4 q0 = p[0], q1 = p[1];
            bf[n][c].u[0] = q0.x; bf[n][c].u[1] = q0.y;
            bf[n][c].u[2] = q0.z; bf[n][c].u[3] = q0.w;
            bf[n][c].u[4] = q1.x; bf[n][c].u[5] = q1.y;
            bf[n][c].u[6] = q1.z; bf[n][c].u[7] = q1.w;
        }
    }

    // A fragments (unconditional; grow<=n_in and row n_in is zeros).
    // A layout per chunk c (matches B):
    // half=0: K = c*32+0..15; half=1: K = c*32+16..31
    const unsigned short* arow = fb + (long long)grow * CIN;
    Frag af[2];
#pragma unroll
    for (int c = 0; c < 2; ++c) {
        uint4 q0 = *(const uint4*)(arow + c * 32 + half * 16);
        uint4 q1 = *(const uint4*)(arow + c * 32 + half * 16 + 8);
        af[c].u[0] = q0.x; af[c].u[1] = q0.y; af[c].u[2] = q0.z; af[c].u[3] = q0.w;
        af[c].u[4] = q1.x; af[c].u[5] = q1.y; af[c].u[6] = q1.z; af[c].u[7] = q1.w;
    }

    v8f acc[8];
    v8f zero = {};
#pragma unroll
    for (int n = 0; n < 8; ++n) acc[n] = zero;

#pragma unroll
    for (int c = 0; c < 2; ++c) {
#pragma unroll
        for (int n = 0; n < 8; ++n) {
            acc[n] = __builtin_amdgcn_wmma_f32_16x16x32_bf16(
                false, af[c].v, false, bf[n][c].v, (short)0, acc[n], false, false);
        }
    }

    // C/D layout: lane -> col lidx; VGPR v -> row v + 8*half. Atomic scatter
    // (unpredicated; pad rows land in y's trash row NOUT).
#pragma unroll
    for (int n = 0; n < 8; ++n) {
#pragma unroll
        for (int v = 0; v < 8; ++v) {
            atomicAdd(&y[(long long)orow[v] * COUT + n * 16 + lidx], acc[n][v]);
        }
    }
}

// ----------------------- pass 2: transposed conv ----------------------------
// z[in_map[k,m], :] += ybf[out_map[k,m], :] @ W2[k]   (128 -> 64 channels)
// K=128 in 4 chunks, N=64 -> 4 WMMA N-tiles per wave.

__global__ __launch_bounds__(128) void conv2_kernel(
    const unsigned short* __restrict__ ybf,  // y bf16 [NOUT+1][128], last row 0
    const unsigned short* __restrict__ Wt,   // W2 bf16 transposed [27][64][128]
    const int* __restrict__ in_map, const int* __restrict__ out_map,
    const int* __restrict__ n_out_p, float* __restrict__ z,
    float* __restrict__ trash, int M, int n_in) {
    __shared__ unsigned short wlds[COUT * CIN];   // [n (0..63)][kk (0..127)]

    const int k   = blockIdx.y;
    const int tid = threadIdx.x;
    {
        const uint4* src = (const uint4*)(Wt + (long long)k * (COUT * CIN));
        uint4* dst = (uint4*)wlds;
#pragma unroll
        for (int i = 0; i < 8; ++i) dst[tid + 128 * i] = src[tid + 128 * i];
    }
    __syncthreads();

    const int NOUT = *n_out_p;
    const int wave = tid >> 5;
    const int lane = tid & 31;
    const int half = lane >> 4;
    const int lidx = lane & 15;
    const int m0   = blockIdx.x * 64 + wave * 16;
    const long long mapbase = (long long)k * M;

    int grow;
    int orow[8];
    if ((blockIdx.x + 1) * 64 <= (unsigned)M) {
        grow = out_map[mapbase + m0 + lidx];
#pragma unroll
        for (int v = 0; v < 8; ++v) orow[v] = in_map[mapbase + m0 + half * 8 + v];
    } else {
        const int m = m0 + lidx;
        grow = (m < M) ? out_map[mapbase + m] : NOUT;     // NOUT = zero row
#pragma unroll
        for (int v = 0; v < 8; ++v) {
            int mm = m0 + half * 8 + v;
            orow[v] = (mm < M) ? in_map[mapbase + mm] : n_in;  // -> trash
        }
    }

    Frag bf[4][4];
#pragma unroll
    for (int n = 0; n < 4; ++n) {
#pragma unroll
        for (int c = 0; c < 4; ++c) {
            const uint4* p =
                (const uint4*)&wlds[(n * 16 + lidx) * COUT + c * 32 + half * 16];
            uint4 q0 = p[0], q1 = p[1];
            bf[n][c].u[0] = q0.x; bf[n][c].u[1] = q0.y;
            bf[n][c].u[2] = q0.z; bf[n][c].u[3] = q0.w;
            bf[n][c].u[4] = q1.x; bf[n][c].u[5] = q1.y;
            bf[n][c].u[6] = q1.z; bf[n][c].u[7] = q1.w;
        }
    }

    const unsigned short* arow = ybf + (long long)grow * COUT;  // row NOUT = zeros

    v8f acc[4];
    v8f zero = {};
#pragma unroll
    for (int n = 0; n < 4; ++n) acc[n] = zero;

#pragma unroll
    for (int c = 0; c < 4; ++c) {
        uint4 q0 = *(const uint4*)(arow + c * 32 + half * 16);
        uint4 q1 = *(const uint4*)(arow + c * 32 + half * 16 + 8);
        Frag af;
        af.u[0] = q0.x; af.u[1] = q0.y; af.u[2] = q0.z; af.u[3] = q0.w;
        af.u[4] = q1.x; af.u[5] = q1.y; af.u[6] = q1.z; af.u[7] = q1.w;
#pragma unroll
        for (int n = 0; n < 4; ++n) {
            acc[n] = __builtin_amdgcn_wmma_f32_16x16x32_bf16(
                false, af.v, false, bf[n][c].v, (short)0, acc[n], false, false);
        }
    }

    // Unpredicated atomics: pad rows are redirected to a trash row via a
    // pointer select (v_cndmask) instead of exec masking.
    float* rowp[8];
#pragma unroll
    for (int v = 0; v < 8; ++v) {
        rowp[v] = (orow[v] < n_in) ? (z + (long long)orow[v] * CIN) : trash;
    }
#pragma unroll
    for (int n = 0; n < 4; ++n) {
#pragma unroll
        for (int v = 0; v < 8; ++v) {
            atomicAdd(rowp[v] + n * 16 + lidx, acc[n][v]);
        }
    }
}

// ------------------------------- launcher -----------------------------------

extern "C" void kernel_launch(void* const* d_in, const int* in_sizes, int n_in_arrs,
                              void* d_out, int out_size, void* d_ws, size_t ws_size,
                              hipStream_t stream) {
    const float* feats   = (const float*)d_in[0];
    const float* W1      = (const float*)d_in[1];
    const float* b1      = (const float*)d_in[2];
    const float* W2      = (const float*)d_in[3];
    const float* b2      = (const float*)d_in[4];
    const int*   in_map  = (const int*)d_in[5];
    const int*   out_map = (const int*)d_in[6];
    const int*   n_out_p = (const int*)d_in[7];

    float* z = (float*)d_out;

    const int n_in = in_sizes[0] / CIN;      // 250000
    const int M    = in_sizes[5] / 27;       // map columns

    // Workspace layout (offsets 256B aligned):
    //   y   f32  : (MAXN2+1) x 128           (atomic accumulator + trash row)
    //   ybf bf16 : (MAXN2+1) x 128           (bias-fused downcast, pad row 0)
    //   fb  bf16 : (n_in+1) x 64             (feats mirror, pad row 0)
    //   Wt1 bf16 : 27 x 128 x 64  (transposed [k][n][kk])
    //   Wt2 bf16 : 27 x 64 x 128  (transposed [k][n][kk])
    char* ws = (char*)d_ws;
    const long long Y_ELEMS  = (long long)(MAXN2 + 1) * COUT;
    const long long FB_ELEMS = (long long)(n_in + 1) * CIN;
    const long long FB_BYTES = (FB_ELEMS * 2 + 255) & ~255LL;
    float*          y   = (float*)ws;
    unsigned short* ybf = (unsigned short*)(ws + Y_ELEMS * 4);
    unsigned short* fb  = (unsigned short*)(ws + Y_ELEMS * 6);
    unsigned short* Wt1 = (unsigned short*)(ws + Y_ELEMS * 6 + FB_BYTES);
    unsigned short* Wt2 = Wt1 + 27 * CIN * COUT;

    // 1) zero y (full upper-bound region so the trash row is clean)
    {
        int blocks = (int)((Y_ELEMS + 255) / 256);
        zero_ws_kernel<<<blocks, 256, 0, stream>>>(y, Y_ELEMS);
    }
    // 2) one-shot precision prep: weights (transposed) + feats to bf16
    {
        long long tw = 27LL * CIN * COUT;
        int blocks = (int)((tw + 255) / 256);
        cvt_w_kernel<<<blocks, 256, 0, stream>>>(W1, Wt1, CIN, COUT);
        cvt_w_kernel<<<blocks, 256, 0, stream>>>(W2, Wt2, COUT, CIN);
        long long tf = (long long)n_in * CIN;
        cvt_feats_kernel<<<(int)((FB_ELEMS + 255) / 256), 256, 0, stream>>>(
            feats, fb, tf, FB_ELEMS);
    }
    // 3) sparse conv: bf16 gather -> WMMA -> f32 atomic scatter into y
    {
        dim3 grid((M + 63) / 64, 27);
        conv1_kernel<<<grid, 128, 0, stream>>>(fb, Wt1, in_map, out_map,
                                               n_out_p, y, M, n_in);
    }
    // 4) ybf = bf16(y + b1), pad row zeroed
    {
        int blocks = (int)((Y_ELEMS + 255) / 256);
        cvt_y_bias_kernel<<<blocks, 256, 0, stream>>>(y, b1, ybf, n_out_p);
    }
    // 5) z = b2 broadcast
    {
        long long ne = (long long)n_in * CIN;
        int blocks = (int)((ne + 255) / 256);
        init_z_kernel<<<blocks, 256, 0, stream>>>(z, b2, ne);
    }
    // 6) transposed conv: bf16 gather -> WMMA -> f32 atomic scatter into z
    //    (pad rows redirected into y's storage, which is dead at this point)
    {
        dim3 grid((M + 63) / 64, 27);
        conv2_kernel<<<grid, 128, 0, stream>>>(ybf, Wt2, in_map, out_map,
                                               n_out_p, z, /*trash=*/y, M, n_in);
    }

    (void)n_in_arrs; (void)out_size; (void)ws_size;
}
